// VanillaAttention_71038759076391
// MI455X (gfx1250) — compile-verified, matrix operand order checked by reading, not run on hardware
//
#include <hip/hip_runtime.h>

typedef __attribute__((ext_vector_type(16))) _Float16 v16h;
typedef __attribute__((ext_vector_type(8)))  _Float16 v8h;
typedef __attribute__((ext_vector_type(8)))  float    v8f;
typedef __attribute__((ext_vector_type(4)))  float    v4f;
typedef __attribute__((ext_vector_type(4)))  int      v4i;

#define S_LEN    2048
#define D_DIM    64
#define KV_BLK   32
#define K_STRIDE  72  // halves: 144B rows (mult of 16B) -> aligned 16B sub-runs
#define KF_STRIDE 68  // floats: 272B rows (fallback kernel)
#define V_STRIDE  48  // halves: 96B rows
#define P_STRIDE  48

// ---- address-space casts via integer route (LDS generic addr low 32 bits == LDS offset) ----
typedef __attribute__((address_space(1))) v4i as1_v4i;
typedef __attribute__((address_space(3))) v4i as3_v4i;
__device__ __forceinline__ as3_v4i* lds_cast(const void* p) {
    return (as3_v4i*)(unsigned)(unsigned long long)p;
}
__device__ __forceinline__ as1_v4i* gbl_cast(const void* p) {
    return (as1_v4i*)(unsigned long long)p;
}

// 16-lane XOR butterfly via v_permlane16_b32 (VALU, frees the LDS pipe)
__device__ __forceinline__ float xor16(float v, unsigned s0, unsigned s1) {
    unsigned u = __float_as_uint(v);
    u = __builtin_amdgcn_permlane16(u, u, s0, s1, false, false);
    return __uint_as_float(u);
}

__device__ __forceinline__ v16h frag_from_f16(const _Float16* p0, const _Float16* p1) {
    v8h lo = *(const v8h*)p0;
    v8h hi = *(const v8h*)p1;
    v16h f;
#pragma unroll
    for (int i = 0; i < 8; ++i) { f[i] = lo[i]; f[i + 8] = hi[i]; }
    return f;
}

__device__ __forceinline__ v16h frag_from_f32(const float* p0, const float* p1) {
    v4f a0 = *(const v4f*)p0, a1 = *(const v4f*)(p0 + 4);
    v4f b0 = *(const v4f*)p1, b1 = *(const v4f*)(p1 + 4);
    v16h f;
#pragma unroll
    for (int i = 0; i < 4; ++i) {
        f[i]      = (_Float16)a0[i];
        f[i + 4]  = (_Float16)a1[i];
        f[i + 8]  = (_Float16)b0[i];
        f[i + 12] = (_Float16)b1[i];
    }
    return f;
}

// XOR-butterfly selects for masks 1,2,4,8 (lane0 in low nibble; selA=lanes0-7, selB=lanes8-15)
#define SEL_A0 0x67452301u
#define SEL_A1 0x54761032u
#define SEL_A2 0x32107654u
#define SEL_A3 0xFEDCBA98u
#define SEL_B0 0xEFCDAB89u
#define SEL_B1 0xDCFE98BAu
#define SEL_B2 0xBA98FEDCu
#define SEL_B3 0x76543210u

// =====================================================================================
// Prepass: K -> f16 row-major [bh][key][d];  V -> f16 transposed [bh][d][key]
// One block per (bh, 64-key tile); LDS tile turns the V transpose into coalesced stores.
// =====================================================================================
__global__ __launch_bounds__(256) void convert_kv_f16(const float* __restrict__ k_g,
                                                      const float* __restrict__ v_g,
                                                      _Float16* __restrict__ kf,
                                                      _Float16* __restrict__ vt)
{
    __shared__ _Float16 tile[64 * 72];           // [key][d], +8 pad
    const int bh = blockIdx.x >> 5;              // 0..63
    const int kt = blockIdx.x & 31;              // 64-key tile
    const float* ks = k_g + ((size_t)bh * S_LEN + kt * 64) * D_DIM;
    const float* vs = v_g + ((size_t)bh * S_LEN + kt * 64) * D_DIM;
    _Float16* kd = kf + ((size_t)bh * S_LEN + kt * 64) * D_DIM;
    _Float16* vd = vt + (size_t)bh * D_DIM * S_LEN + kt * 64;

    const int t   = threadIdx.x;
    const int key = t >> 2;
    const int d0  = (t & 3) * 16;
#pragma unroll
    for (int i = 0; i < 16; ++i) {
        kd[key * D_DIM + d0 + i]  = (_Float16)ks[key * D_DIM + d0 + i];
        tile[key * 72 + d0 + i]   = (_Float16)vs[key * D_DIM + d0 + i];
    }
    __syncthreads();
    const int d   = t >> 2;
    const int kk0 = (t & 3) * 16;
#pragma unroll
    for (int i = 0; i < 16; ++i)
        vd[(size_t)d * S_LEN + kk0 + i] = tile[(kk0 + i) * 72 + d];
}

// =====================================================================================
// Main kernel (fast path): f16 K/V in workspace, double-buffered async-DMA tiles.
// =====================================================================================
__global__ __launch_bounds__(256) void
flash_attn_wmma_ws(const float* __restrict__ q_g,
                   const _Float16* __restrict__ kf_g,   // [bh][key][d] f16
                   const _Float16* __restrict__ vt_g,   // [bh][d][key] f16
                   const int*   __restrict__ mask_g,
                   float* __restrict__ o_g)
{
    __shared__ __align__(16) _Float16 k_lds[2][KV_BLK * K_STRIDE];  // 32 keys x 64 d
    __shared__ __align__(16) _Float16 v_lds[2][D_DIM * V_STRIDE];   // 64 d x 32 keys
    __shared__ __align__(16) _Float16 p_lds[8 * 16 * P_STRIDE];     // per-wave P staging

    const int tid  = threadIdx.x;
    const int lane = tid & 31;
    const int wv   = tid >> 5;
    const int bh   = blockIdx.x >> 4;
    const int qblk = blockIdx.x & 15;
    const int b    = bh >> 4;              // H == 16
    const int q0   = qblk * 128 + wv * 16;

    const float*    qh = q_g  + (size_t)bh * S_LEN * D_DIM;
    const _Float16* kf = kf_g + (size_t)bh * S_LEN * D_DIM;
    const _Float16* vt = vt_g + (size_t)bh * D_DIM * S_LEN;
    float*          oh = o_g  + (size_t)bh * S_LEN * D_DIM;
    const int*      mk = mask_g + b * S_LEN;

    const int n     = lane & 15;
    const int hi    = lane >> 4;
    const int kbase = hi * 8;

    // async chunk mapping: K tile 32x64 halves = 256 x 16B ; V tile 64x32 halves = 256 x 16B
    const int kkey = tid >> 3, kq = (tid & 7) * 8;   // K: 8 halves per chunk along d
    const int vd   = tid >> 2, vq = (tid & 3) * 8;   // V: 8 halves per chunk along key

    // ---- Q A-fragments, pre-scaled by 1/sqrt(D) = 0.125 ----
    v16h qa[2];
#pragma unroll
    for (int f = 0; f < 2; ++f) {
#pragma unroll
        for (int h = 0; h < 16; ++h) {
            int d = kbase + (h & 7) + 16 * (h >> 3) + 32 * f;
            qa[f][h] = (_Float16)(qh[(q0 + n) * D_DIM + d] * 0.125f);
        }
    }

    v8f acc[4];
#pragma unroll
    for (int f = 0; f < 4; ++f) acc[f] = (v8f){};
    float mx[8], l[8];
#pragma unroll
    for (int j = 0; j < 8; ++j) { mx[j] = -1.0e30f; l[j] = 0.0f; }

    // ---- prologue: issue tile 0 into buffer 0 ----
    __builtin_amdgcn_global_load_async_to_lds_b128(
        gbl_cast(kf + (size_t)kkey * D_DIM + kq),
        lds_cast(&k_lds[0][kkey * K_STRIDE + kq]), 0, 0);
    __builtin_amdgcn_global_load_async_to_lds_b128(
        gbl_cast(vt + (size_t)vd * S_LEN + vq),
        lds_cast(&v_lds[0][vd * V_STRIDE + vq]), 0, 0);

    for (int kb = 0; kb < S_LEN; kb += KV_BLK) {
        const int p  = (kb >> 5) & 1;
        const int nx = p ^ 1;

        // ---- issue next tile into the other buffer; wait only for current tile ----
        if (kb + KV_BLK < S_LEN) {
            __builtin_amdgcn_global_load_async_to_lds_b128(
                gbl_cast(kf + (size_t)(kb + KV_BLK + kkey) * D_DIM + kq),
                lds_cast(&k_lds[nx][kkey * K_STRIDE + kq]), 0, 0);
            __builtin_amdgcn_global_load_async_to_lds_b128(
                gbl_cast(vt + (size_t)vd * S_LEN + (kb + KV_BLK) + vq),
                lds_cast(&v_lds[nx][vd * V_STRIDE + vq]), 0, 0);
            asm volatile("s_wait_asynccnt 2" ::: "memory");
        } else {
            asm volatile("s_wait_asynccnt 0" ::: "memory");
        }
        __syncthreads();   // whole current tile visible to all waves

        const _Float16* kt = k_lds[p];
        const _Float16* vtile = v_lds[p];

        // ---- S = Q K^T ----
        v8f s[2];
#pragma unroll
        for (int c = 0; c < 2; ++c) {
            const _Float16* col = &kt[(c * 16 + n) * K_STRIDE + kbase];
            v16h b0 = frag_from_f16(col,      col + 16);
            v16h b1 = frag_from_f16(col + 32, col + 48);
            v8f z = (v8f){};
            z    = __builtin_amdgcn_wmma_f32_16x16x32_f16(false, qa[0], false, b0, (short)0, z, false, false);
            s[c] = __builtin_amdgcn_wmma_f32_16x16x32_f16(false, qa[1], false, b1, (short)0, z, false, false);
        }

        // ---- additive key-padding mask ----
        const float m0 = mk[kb + n]      ? 0.0f : -1.0e6f;
        const float m1 = mk[kb + 16 + n] ? 0.0f : -1.0e6f;
#pragma unroll
        for (int j = 0; j < 8; ++j) { s[0][j] += m0; s[1][j] += m1; }

        // ---- online softmax (permlane16 butterflies) ----
        float t[8], p0[8], p1[8], rs[8];
#pragma unroll
        for (int j = 0; j < 8; ++j) t[j] = fmaxf(s[0][j], s[1][j]);
#pragma unroll
        for (int j = 0; j < 8; ++j) {
            t[j] = fmaxf(t[j], xor16(t[j], SEL_A0, SEL_B0));
            t[j] = fmaxf(t[j], xor16(t[j], SEL_A1, SEL_B1));
            t[j] = fmaxf(t[j], xor16(t[j], SEL_A2, SEL_B2));
            t[j] = fmaxf(t[j], xor16(t[j], SEL_A3, SEL_B3));
        }
#pragma unroll
        for (int j = 0; j < 8; ++j) {
            float nm = fmaxf(mx[j], t[j]);
            float al = __expf(mx[j] - nm);
            mx[j] = nm;
            p0[j] = __expf(s[0][j] - nm);
            p1[j] = __expf(s[1][j] - nm);
            rs[j] = p0[j] + p1[j];
            l[j] *= al;
#pragma unroll
            for (int f = 0; f < 4; ++f) acc[f][j] *= al;
        }
#pragma unroll
        for (int j = 0; j < 8; ++j) {
            rs[j] += xor16(rs[j], SEL_A0, SEL_B0);
            rs[j] += xor16(rs[j], SEL_A1, SEL_B1);
            rs[j] += xor16(rs[j], SEL_A2, SEL_B2);
            rs[j] += xor16(rs[j], SEL_A3, SEL_B3);
            l[j]  += rs[j];
        }

        // ---- stage P (C-layout -> A-layout) via per-wave LDS pad ----
        _Float16* pw = &p_lds[wv * 16 * P_STRIDE];
#pragma unroll
        for (int j = 0; j < 8; ++j) {
            int m = j + 8 * hi;
            pw[m * P_STRIDE + n]      = (_Float16)p0[j];
            pw[m * P_STRIDE + 16 + n] = (_Float16)p1[j];
        }
        asm volatile("s_wait_dscnt 0" ::: "memory");  // DS in-order per wave

        const _Float16* prow = &pw[n * P_STRIDE + kbase];
        v16h pa = frag_from_f16(prow, prow + 16);

        // ---- acc += P V ----
#pragma unroll
        for (int f = 0; f < 4; ++f) {
            const _Float16* vrow = &vtile[(f * 16 + n) * V_STRIDE + kbase];
            v16h vb = frag_from_f16(vrow, vrow + 16);
            acc[f] = __builtin_amdgcn_wmma_f32_16x16x32_f16(false, pa, false, vb, (short)0, acc[f], false, false);
        }
        __syncthreads();   // all waves done with buffer p before it is refilled
    }

#pragma unroll
    for (int j = 0; j < 8; ++j) {
        int m = j + 8 * hi;
        float inv = 1.0f / fmaxf(l[j], 1e-20f);
#pragma unroll
        for (int f = 0; f < 4; ++f)
            oh[(q0 + m) * D_DIM + f * 16 + n] = acc[f][j] * inv;
    }
}

// =====================================================================================
// Fallback (no workspace): f32 K async tile + in-loop conversion (round-3 kernel).
// =====================================================================================
__global__ __launch_bounds__(256) void
flash_attn_wmma_f16(const float* __restrict__ q_g,
                    const float* __restrict__ k_g,
                    const float* __restrict__ v_g,
                    const int*   __restrict__ mask_g,
                    float* __restrict__ o_g)
{
    __shared__ __align__(16) float    k_ldsf[KV_BLK * KF_STRIDE];
    __shared__ __align__(16) _Float16 v_lds[D_DIM * V_STRIDE];
    __shared__ __align__(16) _Float16 p_lds[8 * 16 * P_STRIDE];

    const int tid  = threadIdx.x;
    const int lane = tid & 31;
    const int wv   = tid >> 5;
    const int bh   = blockIdx.x >> 4;
    const int qblk = blockIdx.x & 15;
    const int b    = bh >> 4;
    const int q0   = qblk * 128 + wv * 16;

    const float* qh = q_g + (size_t)bh * S_LEN * D_DIM;
    const float* kh = k_g + (size_t)bh * S_LEN * D_DIM;
    const float* vh = v_g + (size_t)bh * S_LEN * D_DIM;
    float*       oh = o_g + (size_t)bh * S_LEN * D_DIM;
    const int*   mk = mask_g + b * S_LEN;

    const int n     = lane & 15;
    const int hi    = lane >> 4;
    const int kbase = hi * 8;

    const int c0key = tid >> 4,         c0q = tid & 15;
    const int c1key = (tid + 256) >> 4, c1q = (tid + 256) & 15;

    v16h qa[2];
#pragma unroll
    for (int f = 0; f < 2; ++f) {
#pragma unroll
        for (int h = 0; h < 16; ++h) {
            int d = kbase + (h & 7) + 16 * (h >> 3) + 32 * f;
            qa[f][h] = (_Float16)(qh[(q0 + n) * D_DIM + d] * 0.125f);
        }
    }

    v8f acc[4];
#pragma unroll
    for (int f = 0; f < 4; ++f) acc[f] = (v8f){};
    float mx[8], l[8];
#pragma unroll
    for (int j = 0; j < 8; ++j) { mx[j] = -1.0e30f; l[j] = 0.0f; }

    for (int kb = 0; kb < S_LEN; kb += KV_BLK) {
        __syncthreads();

        __builtin_amdgcn_global_load_async_to_lds_b128(
            gbl_cast(kh + (size_t)(kb + c0key) * D_DIM + c0q * 4),
            lds_cast(&k_ldsf[c0key * KF_STRIDE + c0q * 4]), 0, 0);
        __builtin_amdgcn_global_load_async_to_lds_b128(
            gbl_cast(kh + (size_t)(kb + c1key) * D_DIM + c1q * 4),
            lds_cast(&k_ldsf[c1key * KF_STRIDE + c1q * 4]), 0, 0);

        {
            const int key = tid >> 3;
            const int d0  = (tid & 7) * 8;
            const float* vsrc = vh + (size_t)(kb + key) * D_DIM + d0;
#pragma unroll
            for (int i = 0; i < 8; ++i)
                v_lds[(d0 + i) * V_STRIDE + key] = (_Float16)vsrc[i];
        }
        asm volatile("s_wait_asynccnt 0" ::: "memory");
        __syncthreads();

        v8f s[2];
#pragma unroll
        for (int c = 0; c < 2; ++c) {
            const float* colf = &k_ldsf[(c * 16 + n) * KF_STRIDE + kbase];
            v16h b0 = frag_from_f32(colf,      colf + 16);
            v16h b1 = frag_from_f32(colf + 32, colf + 48);
            v8f z = (v8f){};
            z    = __builtin_amdgcn_wmma_f32_16x16x32_f16(false, qa[0], false, b0, (short)0, z, false, false);
            s[c] = __builtin_amdgcn_wmma_f32_16x16x32_f16(false, qa[1], false, b1, (short)0, z, false, false);
        }

        const float m0 = mk[kb + n]      ? 0.0f : -1.0e6f;
        const float m1 = mk[kb + 16 + n] ? 0.0f : -1.0e6f;
#pragma unroll
        for (int j = 0; j < 8; ++j) { s[0][j] += m0; s[1][j] += m1; }

        float t[8], p0[8], p1[8], rs[8];
#pragma unroll
        for (int j = 0; j < 8; ++j) t[j] = fmaxf(s[0][j], s[1][j]);
#pragma unroll
        for (int j = 0; j < 8; ++j) {
            t[j] = fmaxf(t[j], xor16(t[j], SEL_A0, SEL_B0));
            t[j] = fmaxf(t[j], xor16(t[j], SEL_A1, SEL_B1));
            t[j] = fmaxf(t[j], xor16(t[j], SEL_A2, SEL_B2));
            t[j] = fmaxf(t[j], xor16(t[j], SEL_A3, SEL_B3));
        }
#pragma unroll
        for (int j = 0; j < 8; ++j) {
            float nm = fmaxf(mx[j], t[j]);
            float al = __expf(mx[j] - nm);
            mx[j] = nm;
            p0[j] = __expf(s[0][j] - nm);
            p1[j] = __expf(s[1][j] - nm);
            rs[j] = p0[j] + p1[j];
            l[j] *= al;
#pragma unroll
            for (int f = 0; f < 4; ++f) acc[f][j] *= al;
        }
#pragma unroll
        for (int j = 0; j < 8; ++j) {
            rs[j] += xor16(rs[j], SEL_A0, SEL_B0);
            rs[j] += xor16(rs[j], SEL_A1, SEL_B1);
            rs[j] += xor16(rs[j], SEL_A2, SEL_B2);
            rs[j] += xor16(rs[j], SEL_A3, SEL_B3);
            l[j]  += rs[j];
        }

        _Float16* pw = &p_lds[wv * 16 * P_STRIDE];
#pragma unroll
        for (int j = 0; j < 8; ++j) {
            int m = j + 8 * hi;
            pw[m * P_STRIDE + n]      = (_Float16)p0[j];
            pw[m * P_STRIDE + 16 + n] = (_Float16)p1[j];
        }
        asm volatile("s_wait_dscnt 0" ::: "memory");

        const _Float16* prow = &pw[n * P_STRIDE + kbase];
        v16h pa = frag_from_f16(prow, prow + 16);

#pragma unroll
        for (int f = 0; f < 4; ++f) {
            const _Float16* vrow = &v_lds[(f * 16 + n) * V_STRIDE + kbase];
            v16h vb = frag_from_f16(vrow, vrow + 16);
            acc[f] = __builtin_amdgcn_wmma_f32_16x16x32_f16(false, pa, false, vb, (short)0, acc[f], false, false);
        }
    }

#pragma unroll
    for (int j = 0; j < 8; ++j) {
        int m = j + 8 * hi;
        float inv = 1.0f / fmaxf(l[j], 1e-20f);
#pragma unroll
        for (int f = 0; f < 4; ++f)
            oh[(q0 + m) * D_DIM + f * 16 + n] = acc[f][j] * inv;
    }
}

extern "C" void kernel_launch(void* const* d_in, const int* in_sizes, int n_in,
                              void* d_out, int out_size, void* d_ws, size_t ws_size,
                              hipStream_t stream) {
    (void)in_sizes; (void)n_in; (void)out_size;
    const float* q  = (const float*)d_in[0];
    const float* k  = (const float*)d_in[1];
    const float* v  = (const float*)d_in[2];
    const int*   mk = (const int*)d_in[3];
    float* out = (float*)d_out;

    const size_t head_elems = (size_t)64 * S_LEN * D_DIM;          // 8.4M f16 each
    const size_t ws_need    = 2 * head_elems * sizeof(_Float16);   // 33.5 MB

    dim3 grid(1024), block(256);
    if (ws_size >= ws_need) {
        _Float16* kf = (_Float16*)d_ws;
        _Float16* vt = kf + head_elems;
        convert_kv_f16<<<dim3(2048), block, 0, stream>>>(k, v, kf, vt);
        flash_attn_wmma_ws<<<grid, block, 0, stream>>>(q, kf, vt, mk, out);
    } else {
        flash_attn_wmma_f16<<<grid, block, 0, stream>>>(q, k, v, mk, out);
    }
}
